// GATEncoder_6786048328255
// MI455X (gfx1250) — matrix operand order repaired and checked by reading, not compiled, not run on hardware
//
#include <hip/hip_runtime.h>
#include <hip/hip_bf16.h>
#include <math.h>

#define NN 50000
#define NE 800000
#define IN_DIM 128
#define HID 64
#define HEADS 4
#define EMB 128
#define N_GRAPHS 64
#define NEG_SLOPE 0.2f

typedef __attribute__((ext_vector_type(16))) _Float16 v16h;
typedef __attribute__((ext_vector_type(8)))  float    v8f;

__device__ __forceinline__ float lrelu_f(float x) { return x > 0.f ? x : NEG_SLOPE * x; }
__device__ __forceinline__ float elu_f(float x)   { return x > 0.f ? x : __expf(x) - 1.f; }

__device__ __forceinline__ void atomicMaxF(float* addr, float val) {
    unsigned* ua = (unsigned*)addr;
    unsigned old = *(volatile unsigned*)ua;
    while (__uint_as_float(old) < val) {
        unsigned assumed = old;
        old = atomicCAS(ua, assumed, __float_as_uint(val));
        if (old == assumed) break;
    }
}

// ---------------- utility fills ----------------
__global__ void fill_f32_k(float* p, float v, long long n) {
    long long i = (long long)blockIdx.x * blockDim.x + threadIdx.x;
    if (i < n) p[i] = v;
}

// W[K][N] f32 -> Bt[N][K] f16 (transposed, packed for WMMA B-fragment loads)
__global__ void prep_wt_k(const float* __restrict__ W, _Float16* __restrict__ Bt, int K, int N) {
    int i = blockIdx.x * blockDim.x + threadIdx.x;
    if (i < K * N) {
        int k = i / N, n = i % N;
        Bt[(size_t)n * K + k] = (_Float16)W[i];
    }
}

// ---------------- WMMA GEMM: C[M,N] = A[M,K](f32) * Bt[N,K](f16) ----------------
// one wave per 16x64 C tile: one A fragment feeds 4 independent WMMAs per K-step.
__global__ void __launch_bounds__(32) wmma_gemm_k(const float* __restrict__ A,
                                                  const _Float16* __restrict__ Bt,
                                                  float* __restrict__ C,
                                                  int M, int N, int K) {
    const int lane = threadIdx.x;
    const int n0 = blockIdx.x * 64;
    const int m0 = blockIdx.y * 16;
    const int m  = lane & 15;
    const int kh = lane >> 4;           // 0: K-low half of fragment, 1: K-high half

    v8f acc0 = {}, acc1 = {}, acc2 = {}, acc3 = {};
    const float*    arow  = A  + (size_t)(m0 + m) * K;
    const _Float16* bbase = Bt + (size_t)(n0 + m) * K;   // column n = n0 + m (+16 per sub-tile)

    for (int kb = 0; kb < K; kb += 32) {
        __builtin_prefetch(arow + kb + 32, 0, 1);        // global_prefetch_b8
        v16h a;
#pragma unroll
        for (int i = 0; i < 8; ++i) {
            a[i]     = (_Float16)arow[kb + 16 * kh + i];       // K = 16kh..16kh+7
            a[8 + i] = (_Float16)arow[kb + 16 * kh + 8 + i];   // K = 16kh+8..16kh+15
        }
        const int boff = kb + 16 * kh;                   // K = 16kh..16kh+15 within step
        v16h b0 = *(const v16h*)(bbase + (size_t)0  * K + boff);
        v16h b1 = *(const v16h*)(bbase + (size_t)16 * K + boff);
        v16h b2 = *(const v16h*)(bbase + (size_t)32 * K + boff);
        v16h b3 = *(const v16h*)(bbase + (size_t)48 * K + boff);
        acc0 = __builtin_amdgcn_wmma_f32_16x16x32_f16(false, a, false, b0, (short)0, acc0, false, false);
        acc1 = __builtin_amdgcn_wmma_f32_16x16x32_f16(false, a, false, b1, (short)0, acc1, false, false);
        acc2 = __builtin_amdgcn_wmma_f32_16x16x32_f16(false, a, false, b2, (short)0, acc2, false, false);
        acc3 = __builtin_amdgcn_wmma_f32_16x16x32_f16(false, a, false, b3, (short)0, acc3, false, false);
    }
    float* crow = C + (size_t)(m0 + 8 * kh) * N + n0 + m;
#pragma unroll
    for (int r = 0; r < 8; ++r) {
        crow[(size_t)r * N + 0]  = acc0[r];
        crow[(size_t)r * N + 16] = acc1[r];
        crow[(size_t)r * N + 32] = acc2[r];
        crow[(size_t)r * N + 48] = acc3[r];
    }
}

// ---------------- attention logits per (node, head) ----------------
__global__ void alphas_k(const float* __restrict__ H, const float* __restrict__ a_src,
                         const float* __restrict__ a_dst, float* __restrict__ as,
                         float* __restrict__ ad, int N, int heads, int C) {
    int idx = blockIdx.x * blockDim.x + threadIdx.x;
    if (idx >= N * heads) return;
    int n = idx / heads, h = idx % heads;
    const float* hp = H + (size_t)n * heads * C + (size_t)h * C;
    const float* s  = a_src + (size_t)h * C;
    const float* d  = a_dst + (size_t)h * C;
    float vs = 0.f, vd = 0.f;
    for (int c = 0; c < C; ++c) { vs += hp[c] * s[c]; vd += hp[c] * d[c]; }
    as[idx] = vs; ad[idx] = vd;
}

// ---------------- pass A: segment max over dst ----------------
__global__ void edge_max_k(const long long* __restrict__ esrc, const long long* __restrict__ edst,
                           const float* __restrict__ as, const float* __restrict__ ad,
                           float* __restrict__ m, int E0, int N, int heads) {
    int idx = blockIdx.x * blockDim.x + threadIdx.x;
    int ET = E0 + N;                         // +N self loops
    if (idx >= ET * heads) return;
    int e = idx / heads, h = idx % heads;
    int s, d;
    if (e < E0) { s = (int)esrc[e]; d = (int)edst[e]; } else { s = d = e - E0; }
    float ev = lrelu_f(as[s * heads + h] + ad[d * heads + h]);
    atomicMaxF(&m[d * heads + h], ev);
}

// ---------------- pass B: scatter-add of w and w*h (wave per edge) ----------------
template <int HN, int C>
__global__ void edge_agg_k(const long long* __restrict__ esrc, const long long* __restrict__ edst,
                           const float* __restrict__ as, const float* __restrict__ ad,
                           const float* __restrict__ m, const float* __restrict__ H,
                           float* __restrict__ agg, float* __restrict__ z, int E0, int N) {
    constexpr int TOT = HN * C;
    constexpr int CPL = TOT / 32;            // channels per lane (8 or 4) -> one head per lane
    const int lane = threadIdx.x & 31;
    const int wv   = threadIdx.x >> 5;
    long long e = (long long)blockIdx.x * (blockDim.x >> 5) + wv;
    const long long ET = (long long)E0 + N;
    if (e >= ET) return;
    int s, d;
    if (e < E0) { s = (int)esrc[e]; d = (int)edst[e]; } else { s = d = (int)(e - E0); }

    const int c0 = lane * CPL;
    const int h  = c0 / C;
    float ev = lrelu_f(as[s * HN + h] + ad[d * HN + h]);
    float w  = __expf(ev - m[d * HN + h]);

    const float* hs = H   + (size_t)s * TOT;
    float*       ag = agg + (size_t)d * TOT;
#pragma unroll
    for (int j = 0; j < CPL; ++j)
        atomicAdd(&ag[c0 + j], w * hs[c0 + j]);
    if ((c0 % C) == 0)
        atomicAdd(&z[(size_t)d * HN + h], w);
}

// ---------------- per-node epilogue: normalize + bias + elu ----------------
__global__ void node_post_k(const float* __restrict__ agg, const float* __restrict__ z,
                            const float* __restrict__ bias, float* __restrict__ out,
                            int N, int heads, int C) {
    const int tot = heads * C;
    long long i = (long long)blockIdx.x * blockDim.x + threadIdx.x;
    if (i >= (long long)N * tot) return;
    int n = (int)(i / tot), c = (int)(i % tot), h = c / C;
    out[i] = elu_f(agg[i] / z[(size_t)n * heads + h] + bias[c]);
}

// ---------------- global mean pool ----------------
__global__ void pool_acc_k(const float* __restrict__ h, const long long* __restrict__ batch,
                           float* __restrict__ psum, float* __restrict__ pcnt, int N) {
    long long i = (long long)blockIdx.x * blockDim.x + threadIdx.x;
    if (i >= (long long)N * EMB) return;
    int n = (int)(i / EMB), c = (int)(i % EMB);
    int g = (int)batch[n];
    atomicAdd(&psum[(size_t)g * EMB + c], h[i]);
    if (c == 0) atomicAdd(&pcnt[g], 1.f);
}

__global__ void pool_fin_k(const float* __restrict__ psum, const float* __restrict__ pcnt,
                           float* __restrict__ out) {
    int i = blockIdx.x * blockDim.x + threadIdx.x;
    if (i < N_GRAPHS * EMB) out[i] = psum[i] / fmaxf(pcnt[i / EMB], 1.f);
}

// =====================================================================
extern "C" void kernel_launch(void* const* d_in, const int* in_sizes, int n_in,
                              void* d_out, int out_size, void* d_ws, size_t ws_size,
                              hipStream_t stream) {
    const float*     x      = (const float*)d_in[0];
    const long long* eidx   = (const long long*)d_in[1];   // [2, NE] int64
    const long long* batch  = (const long long*)d_in[3];   // [NN] int64 (edge_weight d_in[2] unused)
    const float*     W1     = (const float*)d_in[4];
    const float*     a_src1 = (const float*)d_in[5];
    const float*     a_dst1 = (const float*)d_in[6];
    const float*     b1     = (const float*)d_in[7];
    const float*     W2     = (const float*)d_in[8];
    const float*     a_src2 = (const float*)d_in[9];
    const float*     a_dst2 = (const float*)d_in[10];
    const float*     b2     = (const float*)d_in[11];
    float*           out    = (float*)d_out;

    const long long* esrc = eidx;
    const long long* edst = eidx + NE;

    // ---- workspace carve ----
    char* ws = (char*)d_ws;
    size_t off = 0;
    auto carve = [&](size_t bytes) -> char* {
        char* p = ws + off;
        off += (bytes + 255) & ~(size_t)255;
        return p;
    };
    float*     bufA = (float*)carve((size_t)NN * 256 * 4);   // h1 -> h1_post
    float*     bufB = (float*)carve((size_t)NN * 256 * 4);   // agg1 -> h2
    float*     bufC = (float*)carve((size_t)NN * 128 * 4);   // agg2 -> h2_post
    float*     as1  = (float*)carve((size_t)NN * HEADS * 4);
    float*     ad1  = (float*)carve((size_t)NN * HEADS * 4);
    float*     m1   = (float*)carve((size_t)NN * HEADS * 4);
    float*     z1   = (float*)carve((size_t)NN * HEADS * 4);
    float*     as2  = (float*)carve((size_t)NN * 4);
    float*     ad2  = (float*)carve((size_t)NN * 4);
    float*     m2   = (float*)carve((size_t)NN * 4);
    float*     z2   = (float*)carve((size_t)NN * 4);
    _Float16*  W1t  = (_Float16*)carve((size_t)IN_DIM * HEADS * HID * 2);
    _Float16*  W2t  = (_Float16*)carve((size_t)HEADS * HID * EMB * 2);
    float*     psum = (float*)carve((size_t)N_GRAPHS * EMB * 4);
    float*     pcnt = (float*)carve((size_t)N_GRAPHS * 4);

    const int B = 256;
    auto blocks = [](long long n, int b) { return (unsigned)((n + b - 1) / b); };
    const int ET = NE + NN;

    // ---- weight prepack (f32 -> transposed f16) ----
    prep_wt_k<<<blocks(IN_DIM * HEADS * HID, B), B, 0, stream>>>(W1, W1t, IN_DIM, HEADS * HID);
    prep_wt_k<<<blocks(HEADS * HID * EMB, B), B, 0, stream>>>(W2, W2t, HEADS * HID, EMB);

    // ================= layer 1 =================
    { dim3 g(256 / 64, NN / 16);
      wmma_gemm_k<<<g, 32, 0, stream>>>(x, W1t, bufA, NN, 256, IN_DIM); }

    alphas_k<<<blocks((long long)NN * HEADS, B), B, 0, stream>>>(bufA, a_src1, a_dst1, as1, ad1, NN, HEADS, HID);

    fill_f32_k<<<blocks((long long)NN * HEADS, B), B, 0, stream>>>(m1, -INFINITY, (long long)NN * HEADS);
    fill_f32_k<<<blocks((long long)NN * HEADS, B), B, 0, stream>>>(z1, 0.f, (long long)NN * HEADS);
    fill_f32_k<<<blocks((long long)NN * 256, B), B, 0, stream>>>(bufB, 0.f, (long long)NN * 256);

    edge_max_k<<<blocks((long long)ET * HEADS, B), B, 0, stream>>>(esrc, edst, as1, ad1, m1, NE, NN, HEADS);
    edge_agg_k<HEADS, HID><<<blocks((long long)ET * 32, B), B, 0, stream>>>(esrc, edst, as1, ad1, m1, bufA, bufB, z1, NE, NN);

    node_post_k<<<blocks((long long)NN * 256, B), B, 0, stream>>>(bufB, z1, b1, bufA, NN, HEADS, HID);

    // ================= layer 2 =================
    { dim3 g(128 / 64, NN / 16);
      wmma_gemm_k<<<g, 32, 0, stream>>>(bufA, W2t, bufB, NN, EMB, HEADS * HID); }

    alphas_k<<<blocks((long long)NN, B), B, 0, stream>>>(bufB, a_src2, a_dst2, as2, ad2, NN, 1, EMB);

    fill_f32_k<<<blocks((long long)NN, B), B, 0, stream>>>(m2, -INFINITY, (long long)NN);
    fill_f32_k<<<blocks((long long)NN, B), B, 0, stream>>>(z2, 0.f, (long long)NN);
    fill_f32_k<<<blocks((long long)NN * 128, B), B, 0, stream>>>(bufC, 0.f, (long long)NN * 128);

    edge_max_k<<<blocks((long long)ET, B), B, 0, stream>>>(esrc, edst, as2, ad2, m2, NE, NN, 1);
    edge_agg_k<1, EMB><<<blocks((long long)ET * 32, B), B, 0, stream>>>(esrc, edst, as2, ad2, m2, bufB, bufC, z2, NE, NN);

    node_post_k<<<blocks((long long)NN * 128, B), B, 0, stream>>>(bufC, z2, b2, bufC, NN, 1, EMB);

    // ================= pooling =================
    fill_f32_k<<<blocks(N_GRAPHS * EMB, B), B, 0, stream>>>(psum, 0.f, N_GRAPHS * EMB);
    fill_f32_k<<<1, N_GRAPHS, 0, stream>>>(pcnt, 0.f, N_GRAPHS);
    pool_acc_k<<<blocks((long long)NN * EMB, B), B, 0, stream>>>(bufC, batch, psum, pcnt, NN);
    pool_fin_k<<<blocks(N_GRAPHS * EMB, B), B, 0, stream>>>(psum, pcnt, out);
}
